// CTRNN_63153199120831
// MI455X (gfx1250) — matrix operand order repaired and checked by reading, not compiled, not run on hardware
//
#include <hip/hip_runtime.h>
#include <hip/hip_bf16.h>
#include <math.h>

typedef __attribute__((ext_vector_type(16))) __bf16 v16bf;
typedef __attribute__((ext_vector_type(8)))  __bf16 v8bf;
typedef __attribute__((ext_vector_type(8)))  float  v8f;

#define H      1024
#define DIN    512
#define DOUT   256
#define BATCH  4096
#define BM     32          // batch rows per workgroup (2 M-tiles -> 2x B reuse)
#define LDK    1056        // padded K stride for dynamics (1025 -> 1056, mult of 32)
#define NWAVES 32
#define BLOCK  (NWAVES * 32)
#define NSTEPS 32

__device__ __host__ __forceinline__ __bf16 f2bf(float f) {
  union { float f; unsigned u; } c; c.f = f;
  unsigned r = c.u + 0x7FFFu + ((c.u >> 16) & 1u);   // round-to-nearest-even
  unsigned short h = (unsigned short)(r >> 16);
  __bf16 b; __builtin_memcpy(&b, &h, 2);
  return b;
}

__device__ __forceinline__ v8f vzero() { v8f z = {}; return z; }

// A-fragment: contiguous K order, same as load_b — lanes 0-15 hold K {0..15}, lanes 16-31 K {16..31}
__device__ __forceinline__ void load_a(const __bf16* __restrict__ As, int kt, int mt,
                                       int row, int hi, v16bf& a) {
  const __bf16* aptr = As + (mt * 16 + row) * LDK + kt * 32 + hi * 16;
  v8bf a0 = *(const v8bf*)(aptr);
  v8bf a1 = *(const v8bf*)(aptr + 8);
  a = __builtin_shufflevector(a0, a1, 0,1,2,3,4,5,6,7,8,9,10,11,12,13,14,15);
}

// B-fragment: lane = column N, 16 contiguous K per lane-half (32B aligned, [N][K] weights)
__device__ __forceinline__ void load_b(const __bf16* __restrict__ Bt, int ldb, int kt,
                                       int n, int hi, v16bf& b) {
  b = *(const v16bf*)(Bt + (size_t)n * ldb + kt * 32 + hi * 16);
}

// Double-buffered bf16 GEMM: A (MT 16-row tiles from LDS) x B ([N][K] from global/L2)
// -> MT x NT f32 16x16 accumulators. kt+1 loads are issued before kt's WMMAs so
// global latency hides under the matrix pipe instead of s_wait_loadcnt 0.
template<int MT, int NT, int KTILES>
__device__ __forceinline__ void gemm_bf16(const __bf16* __restrict__ As,
                                          const __bf16* __restrict__ Bt,
                                          int ldb, int nbase, int lane,
                                          v8f acc[MT][NT]) {
  const int row = lane & 15;
  const int hi  = lane >> 4;
  v16bf aF[2][MT], bF[2][NT];
  #pragma unroll
  for (int mt = 0; mt < MT; ++mt) load_a(As, 0, mt, row, hi, aF[0][mt]);
  #pragma unroll
  for (int nt = 0; nt < NT; ++nt) load_b(Bt, ldb, 0, nbase + nt * 16 + row, hi, bF[0][nt]);
  #pragma unroll
  for (int kt = 0; kt < KTILES; ++kt) {
    const int cur = kt & 1, nxt = cur ^ 1;
    if (kt + 1 < KTILES) {
      #pragma unroll
      for (int mt = 0; mt < MT; ++mt) load_a(As, kt + 1, mt, row, hi, aF[nxt][mt]);
      #pragma unroll
      for (int nt = 0; nt < NT; ++nt) load_b(Bt, ldb, kt + 1, nbase + nt * 16 + row, hi, bF[nxt][nt]);
    }
    #pragma unroll
    for (int mt = 0; mt < MT; ++mt)
      #pragma unroll
      for (int nt = 0; nt < NT; ++nt)
        acc[mt][nt] = __builtin_amdgcn_wmma_f32_16x16x32_bf16(
            false, aF[cur][mt], false, bF[cur][nt], (short)0, acc[mt][nt], false, false);
  }
}

// ---- preprocessing: transpose+convert weights to bf16 [N][ldk], zero-pad K ----
__global__ void transpose_to_bf16(const float* __restrict__ W,   // [K][N] row-major
                                  __bf16* __restrict__ Wt,       // [N][ldk]
                                  int K, int N, int ldk) {
  int idx = blockIdx.x * blockDim.x + threadIdx.x;
  int total = N * ldk;
  if (idx >= total) return;
  int n = idx / ldk;
  int k = idx - n * ldk;
  float v = (k < K) ? W[(size_t)k * N + n] : 0.0f;
  Wt[idx] = f2bf(v);
}

__global__ void neg_inv_kernel(const float* __restrict__ tau, float* __restrict__ nit, int n) {
  int i = blockIdx.x * blockDim.x + threadIdx.x;
  if (i < n) nit[i] = -1.0f / tau[i];
}

// ---- fused CTRNN kernel: state proj + 32 RK4 steps + output proj ----
__global__ void __launch_bounds__(BLOCK) ctrnn_main(
    const float* __restrict__ x,        // [B][DIN]
    const float* __restrict__ b_state,  // [H]
    const float* __restrict__ b_dyn,    // [H]
    const float* __restrict__ b_out,    // [DOUT]
    const float* __restrict__ nitau,    // [H]  = -1/tau
    const __bf16* __restrict__ Wst,     // [H][DIN]
    const __bf16* __restrict__ Wdyn,    // [H][LDK] (rows >= 1025 are zero)
    const __bf16* __restrict__ Wout,    // [DOUT][H]
    float* __restrict__ out)            // [B][DOUT]
{
  __shared__ __bf16 Alds[BM * LDK];     // 66 KB: current RK4 input (bf16) / A operand
  const int tid  = threadIdx.x;
  const int wave = tid >> 5;
  const int lane = tid & 31;
  const int row  = lane & 15;
  const int hi   = lane >> 4;
  const int row0 = blockIdx.x * BM;

  // Stage 0: x tile -> LDS bf16 (cols [0,DIN)); zero-fill padding cols so the
  // K=1056 dynamics GEMM never multiplies garbage (pad weight rows are zero too).
  for (int i = tid; i < BM * LDK; i += BLOCK) {
    int m = i / LDK;
    int k = i - m * LDK;
    float v = (k < DIN) ? x[(size_t)(row0 + m) * DIN + k] : 0.0f;
    Alds[i] = f2bf(v);
  }
  __syncthreads();

  // Each wave owns a 32x32 output tile: state cols [nbase, nbase+32).
  // C-layout element: m = mt*16 + hi*8 + r, n = nbase + nt*16 + row.
  const int nbase = wave * 32;
  float bst[2], bdy[2], nit[2];
  #pragma unroll
  for (int nt = 0; nt < 2; ++nt) {
    int n = nbase + nt * 16 + row;
    bst[nt] = b_state[n];
    bdy[nt] = b_dyn[n];
    nit[nt] = nitau[n];
  }

  // Stage 1: state = tanh(x @ Wst + b_state)
  v8f acc[2][2];
  #pragma unroll
  for (int mt = 0; mt < 2; ++mt)
    #pragma unroll
    for (int nt = 0; nt < 2; ++nt) acc[mt][nt] = vzero();
  gemm_bf16<2, 2, DIN / 32>(Alds, Wst, DIN, nbase, lane, acc);

  v8f y[2][2];
  #pragma unroll
  for (int mt = 0; mt < 2; ++mt)
    #pragma unroll
    for (int nt = 0; nt < 2; ++nt)
      #pragma unroll
      for (int r = 0; r < 8; ++r)
        y[mt][nt][r] = tanhf(acc[mt][nt][r] + bst[nt]);

  __syncthreads();  // done reading x tile
  #pragma unroll
  for (int mt = 0; mt < 2; ++mt)
    #pragma unroll
    for (int nt = 0; nt < 2; ++nt)
      #pragma unroll
      for (int r = 0; r < 8; ++r)
        Alds[(mt * 16 + hi * 8 + r) * LDK + nbase + nt * 16 + row] = f2bf(y[mt][nt][r]);
  if (tid < BM) Alds[tid * LDK + H] = f2bf(0.0f);  // time feature t=0

  const float dt = 1.0f / NSTEPS;
  v8f kprev[2][2], ksum[2][2];
  #pragma unroll
  for (int mt = 0; mt < 2; ++mt)
    #pragma unroll
    for (int nt = 0; nt < 2; ++nt) kprev[mt][nt] = vzero();

  // Stage 2: 32 RK4 steps, 4 dynamics evals each; state stays in VGPRs,
  // eval input round-trips only through LDS (bf16).
  #pragma unroll 1
  for (int s = 0; s < NSTEPS; ++s) {
    const float t = s * dt;
    #pragma unroll
    for (int mt = 0; mt < 2; ++mt)
      #pragma unroll
      for (int nt = 0; nt < 2; ++nt) ksum[mt][nt] = vzero();

    #pragma unroll
    for (int e = 0; e < 4; ++e) {
      const float cin = (e == 0) ? 0.0f : (e == 3 ? 1.0f : 0.5f);  // y' = y + cin*dt*kprev
      const float wgt = (e == 1 || e == 2) ? 2.0f : 1.0f;          // RK4 weight

      __syncthreads();  // LDS A tile (y', t) ready for everyone
      v8f a4[2][2];
      #pragma unroll
      for (int mt = 0; mt < 2; ++mt)
        #pragma unroll
        for (int nt = 0; nt < 2; ++nt) a4[mt][nt] = vzero();
      gemm_bf16<2, 2, LDK / 32>(Alds, Wdyn, LDK, nbase, lane, a4);

      v8f k[2][2];
      #pragma unroll
      for (int mt = 0; mt < 2; ++mt)
        #pragma unroll
        for (int nt = 0; nt < 2; ++nt)
          #pragma unroll
          for (int r = 0; r < 8; ++r) {
            float yp = y[mt][nt][r] + cin * dt * kprev[mt][nt][r];        // f32-exact y'
            float kv = tanhf(a4[mt][nt][r] + bdy[nt]) + yp * nit[nt];     // dynamics + leak
            k[mt][nt][r] = kv;
            ksum[mt][nt][r] += wgt * kv;
          }

      __syncthreads();  // all waves finished reading current LDS A tile
      if (e < 3) {
        const float cn = (e == 2) ? 1.0f : 0.5f;
        #pragma unroll
        for (int mt = 0; mt < 2; ++mt)
          #pragma unroll
          for (int nt = 0; nt < 2; ++nt)
            #pragma unroll
            for (int r = 0; r < 8; ++r)
              Alds[(mt * 16 + hi * 8 + r) * LDK + nbase + nt * 16 + row] =
                  f2bf(y[mt][nt][r] + cn * dt * k[mt][nt][r]);
        if (tid < BM) Alds[tid * LDK + H] = f2bf(t + cn * dt);
      } else {
        // finalize step: y += dt/6 * (k1 + 2k2 + 2k3 + k4); publish for next step
        #pragma unroll
        for (int mt = 0; mt < 2; ++mt)
          #pragma unroll
          for (int nt = 0; nt < 2; ++nt)
            #pragma unroll
            for (int r = 0; r < 8; ++r) {
              y[mt][nt][r] += (dt / 6.0f) * ksum[mt][nt][r];
              Alds[(mt * 16 + hi * 8 + r) * LDK + nbase + nt * 16 + row] = f2bf(y[mt][nt][r]);
            }
        if (tid < BM) Alds[tid * LDK + H] = f2bf(t + dt);
      }
      #pragma unroll
      for (int mt = 0; mt < 2; ++mt)
        #pragma unroll
        for (int nt = 0; nt < 2; ++nt) kprev[mt][nt] = k[mt][nt];
    }
  }

  __syncthreads();  // final hidden state (bf16) in LDS

  // Stage 3: out = hidden @ Wout + b_out. 32 waves x one 16x16 tile:
  // wave w computes rows [16*(w>>4)..), output cols [16*(w&15)..).
  const int mt_o = wave >> 4;
  const int nt_o = wave & 15;
  v8f oacc[1][1];
  oacc[0][0] = vzero();
  gemm_bf16<1, 1, H / 32>(Alds + mt_o * 16 * LDK, Wout, H, nt_o * 16, lane, oacc);
  const int n = nt_o * 16 + row;
  const float bo = b_out[n];
  #pragma unroll
  for (int r = 0; r < 8; ++r)
    out[(size_t)(row0 + mt_o * 16 + hi * 8 + r) * DOUT + n] = oacc[0][0][r] + bo;
}

extern "C" void kernel_launch(void* const* d_in, const int* in_sizes, int n_in,
                              void* d_out, int out_size, void* d_ws, size_t ws_size,
                              hipStream_t stream) {
  (void)in_sizes; (void)n_in; (void)out_size; (void)ws_size;
  const float* x       = (const float*)d_in[0];
  const float* W_state = (const float*)d_in[1];   // [512][1024]
  const float* b_state = (const float*)d_in[2];
  const float* W_dyn   = (const float*)d_in[3];   // [1025][1024]
  const float* b_dyn   = (const float*)d_in[4];
  const float* W_out   = (const float*)d_in[5];   // [1024][256]
  const float* b_out   = (const float*)d_in[6];
  const float* tau     = (const float*)d_in[7];   // [1024]

  // workspace layout (bf16 transposed weights + -1/tau): ~3.74 MB, L2-resident
  __bf16* Wst  = (__bf16*)d_ws;                         // H * DIN
  __bf16* Wdyn = Wst  + (size_t)H * DIN;                // H * LDK (K-padded with zeros)
  __bf16* Wout = Wdyn + (size_t)H * LDK;                // DOUT * H
  float*  nit  = (float*)(Wout + (size_t)DOUT * H);     // H

  {
    int total = H * DIN;
    transpose_to_bf16<<<(total + 255) / 256, 256, 0, stream>>>(W_state, Wst, DIN, H, DIN);
  }
  {
    int total = H * LDK;
    transpose_to_bf16<<<(total + 255) / 256, 256, 0, stream>>>(W_dyn, Wdyn, H + 1, H, LDK);
  }
  {
    int total = DOUT * H;
    transpose_to_bf16<<<(total + 255) / 256, 256, 0, stream>>>(W_out, Wout, H, DOUT, H);
  }
  neg_inv_kernel<<<(H + 255) / 256, 256, 0, stream>>>(tau, nit, H);

  ctrnn_main<<<BATCH / BM, BLOCK, 0, stream>>>(
      x, b_state, b_dyn, b_out, nit, Wst, Wdyn, Wout, (float*)d_out);
}
